// SimMatrixLoss_22299470201233
// MI455X (gfx1250) — matrix operand order repaired and checked by reading, not compiled, not run on hardware
//
#include <hip/hip_runtime.h>
#include <hip/hip_bf16.h>
#include <stddef.h>
#include <stdint.h>

// Problem constants (from reference setup_inputs)
#define NB 64
#define NS 1024
#define DD 256

// GEMM tiling
#define BM 128
#define BN 128
#define BK 32
#define LDSW 48   // ushort stride per row: 32 data + 16 pad -> 96B rows, 16B aligned

// Workspace layout (floats): [0]=denom, [1..64]=batch acc, [256..]=row sq norms,
// then bf16 mirror of d at byte offset DBF_OFF.
#define SQ_OFF 256
#define DBF_OFF ((size_t)(SQ_OFF * 4 + NB * NS * 4))            // 263168 B, 16B aligned
#define WS_NEED (DBF_OFF + (size_t)NB * NS * DD * 2)            // ~33.8 MB

#if defined(__HIP_DEVICE_COMPILE__) && __has_builtin(__builtin_amdgcn_global_load_async_to_lds_b128)
#define HAVE_ASYNC 1
#else
#define HAVE_ASYNC 0
#endif

#define AS1 __attribute__((address_space(1)))
#define AS3 __attribute__((address_space(3)))

typedef __attribute__((__vector_size__(4 * sizeof(int)))) int v4i_vs;

typedef __attribute__((ext_vector_type(16))) __bf16 v16bf;
typedef __attribute__((ext_vector_type(8)))  float  v8f;

union FragU { v16bf v; uint4 q[2]; };

struct __align__(8) US4 { unsigned short x, y, z, w; };

__device__ __forceinline__ unsigned short f2bf(float f) {
  // round-to-nearest-even f32 -> bf16
  unsigned u = __float_as_uint(f);
  u += 0x7FFFu + ((u >> 16) & 1u);
  return (unsigned short)(u >> 16);
}

__device__ __forceinline__ void wait_async0() {
#if HAVE_ASYNC
#if __has_builtin(__builtin_amdgcn_s_wait_asynccnt)
  __builtin_amdgcn_s_wait_asynccnt(0);
#else
  asm volatile("s_wait_asynccnt 0" ::: "memory");
#endif
#endif
}

__device__ __forceinline__ float wave_reduce(float v) {
#pragma unroll
  for (int off = 16; off > 0; off >>= 1) v += __shfl_xor(v, off, 32);
  return v;
}

// ---------------- init: zero scalar accumulators in workspace ----------------
__global__ void init_ws(float* ws) {
  if (threadIdx.x < 128) ws[threadIdx.x] = 0.f;
}

// ---------------- one-time f32 -> bf16 mirror of d_vectors ----------------
__global__ __launch_bounds__(256) void conv_bf16(const float* __restrict__ d,
                                                 unsigned short* __restrict__ dbf) {
  size_t i = ((size_t)blockIdx.x * blockDim.x + threadIdx.x) * 8;
  float4 a = *(const float4*)(d + i);
  float4 c = *(const float4*)(d + i + 4);
  US4 o0 = {f2bf(a.x), f2bf(a.y), f2bf(a.z), f2bf(a.w)};
  US4 o1 = {f2bf(c.x), f2bf(c.y), f2bf(c.z), f2bf(c.w)};
  *(US4*)(dbf + i) = o0;
  *(US4*)(dbf + i + 4) = o1;
}

// ---------------- per-row squared norms: one wave32 per row ----------------
__global__ __launch_bounds__(256) void row_sq(const float* __restrict__ d,
                                              float* __restrict__ sq) {
  int wave = (int)((blockIdx.x * blockDim.x + threadIdx.x) >> 5); // 0..NB*NS-1
  int lane = threadIdx.x & 31;
  const float4* p = (const float4*)(d + (size_t)wave * DD);
  float4 a = p[lane];
  float4 b = p[lane + 32];
  float s = a.x * a.x + a.y * a.y + a.z * a.z + a.w * a.w +
            b.x * b.x + b.y * b.y + b.z * b.z + b.w * b.w;
  s = wave_reduce(s);
  if (lane == 0) sq[wave] = s;
}

// ---------------- denom = sum((W - I)^2) ----------------
__global__ __launch_bounds__(256) void denom_k(const float* __restrict__ W,
                                               float* __restrict__ ws) {
  int tid = (int)(blockIdx.x * blockDim.x + threadIdx.x);
  int stride = (int)(gridDim.x * blockDim.x);
  float s = 0.f;
  for (int i = tid; i < NS * NS; i += stride) {
    int r = i >> 10, c = i & (NS - 1);
    float v = W[i] - (r == c ? 1.f : 0.f);
    s += v * v;
  }
  s = wave_reduce(s);
  if ((threadIdx.x & 31) == 0) atomicAdd(ws, s);
}

// ---------------- shared epilogue: dist -> exp -> masked diff -> partial ----
__device__ __forceinline__ void epilogue(
    v8f acc[2][4], int b, int rowBase, int colBase, int wm, int wn, int ln,
    int h, int lane, const float* __restrict__ S, const float* __restrict__ W,
    const float* __restrict__ sq, float* __restrict__ batchAcc) {
  const float* sqB = sq + (size_t)b * NS;
  float part = 0.f;
#pragma unroll
  for (int tm = 0; tm < 2; ++tm) {
#pragma unroll
    for (int tn = 0; tn < 4; ++tn) {
      int gn = colBase + wn * 64 + tn * 16 + ln;
      float sqn = sqB[gn];
#pragma unroll
      for (int r = 0; r < 8; ++r) {
        int gm = rowBase + wm * 32 + tm * 16 + r + 8 * h;
        float g = acc[tm][tn][r];
        float dist2 = fmaxf(sqB[gm] + sqn - 2.f * g, 0.f);
        bool diag = (gm == gn);
        float dist = sqrtf(dist2 + (diag ? 1.f : 0.f));
        float kd = __expf(-dist);
        float kdt = diag ? 0.f : kd;
        size_t sidx = (size_t)gm * NS + gn;
        float st = S[sidx] - (diag ? 1.f : 0.f);
        float diffv = W[sidx] * (kdt - st);
        part = fmaf(diffv, diffv, part);
      }
    }
  }
  part = wave_reduce(part);
  if (lane == 0) atomicAdd(&batchAcc[b], part);
}

// ---------------- main path: bf16-mirror GEMM, async-to-LDS, double buffered --
__global__ __launch_bounds__(256) void gram_loss_bf16(
    const unsigned short* __restrict__ dbf, const float* __restrict__ S,
    const float* __restrict__ W, const float* __restrict__ sq,
    float* __restrict__ batchAcc) {
  __shared__ __align__(16) unsigned short lA[2][BM * LDSW];
  __shared__ __align__(16) unsigned short lB[2][BN * LDSW];

  const int b = blockIdx.z;
  const int rowBase = blockIdx.y * BM;
  const int colBase = blockIdx.x * BN;
  const int tid = threadIdx.x;
  const int lane = tid & 31;
  const int waveId = tid >> 5;
  const int wm = waveId & 3;   // row group (32 rows)
  const int wn = waveId >> 2;  // col group (64 cols)
  const int ln = lane & 15;
  const int h = lane >> 4;

  const unsigned short* dbfB = dbf + (size_t)b * NS * DD;

  v8f acc[2][4];
#pragma unroll
  for (int i = 0; i < 2; ++i)
#pragma unroll
    for (int j = 0; j < 4; ++j)
#pragma unroll
      for (int e = 0; e < 8; ++e) acc[i][j][e] = 0.f;

  // Stage one K-slice (A:128x32 bf16, B:128x32 bf16) into LDS buffer `buf`.
  // 1024 x 16B transfers; 4 per thread.
  auto stage = [&](int buf, int k0) {
#pragma unroll
    for (int t = 0; t < 4; ++t) {
      int idx = tid + t * 256;       // 0..1023
      int isB = idx >> 9;            // 0 = A tile, 1 = B tile
      int rem = idx & 511;
      int r = rem >> 2;              // 0..127
      int q = rem & 3;               // 16B quarter of the 64B row
      const unsigned short* g =
          dbfB + (size_t)((isB ? colBase : rowBase) + r) * DD + k0 + q * 8;
      unsigned short* l = (isB ? lB[buf] : lA[buf]) + r * LDSW + q * 8;
#if HAVE_ASYNC
      __builtin_amdgcn_global_load_async_to_lds_b128(
          (AS1 v4i_vs*)g, (AS3 v4i_vs*)l, 0, 0);
#else
      *(uint4*)l = *(const uint4*)g;
#endif
    }
  };

  stage(0, 0);
  wait_async0();
  __syncthreads();

  const int NI = DD / BK;  // 8 K-steps
  int buf = 0;
  for (int i = 0; i < NI; ++i) {
    if (i + 1 < NI) stage(buf ^ 1, (i + 1) * BK);  // prefetch next slice

    // ---- B fragments (32x16 bf16): lane holds col n = ln, K run [16h,16h+16)
    FragU bfr[4];
#pragma unroll
    for (int tn = 0; tn < 4; ++tn) {
      int n = wn * 64 + tn * 16 + ln;
      const unsigned short* pb = lB[buf] + n * LDSW + h * 16;
      bfr[tn].q[0] = *(const uint4*)(pb);
      bfr[tn].q[1] = *(const uint4*)(pb + 8);
    }
    // ---- A fragments (16x32 bf16): lane row m = ln; K run [16h,16h+16)
#pragma unroll
    for (int tm = 0; tm < 2; ++tm) {
      int m = wm * 32 + tm * 16 + ln;
      const unsigned short* pa = lA[buf] + m * LDSW;
      FragU af;
      af.q[0] = *(const uint4*)(pa + h * 16);
      af.q[1] = *(const uint4*)(pa + h * 16 + 8);
#pragma unroll
      for (int tn = 0; tn < 4; ++tn) {
        acc[tm][tn] = __builtin_amdgcn_wmma_f32_16x16x32_bf16(
            false, af.v, false, bfr[tn].v, (short)0, acc[tm][tn], false, false);
      }
    }

    wait_async0();     // our async fills done (ASYNCcnt==0)
    __syncthreads();   // everyone's fills + our LDS reads ordered
    buf ^= 1;
  }

  epilogue(acc, b, rowBase, colBase, wm, wn, ln, h, lane, S, W, sq, batchAcc);
}

// ---------------- fallback path (small ws): convert in staging ----------------
__global__ __launch_bounds__(256) void gram_loss_f32(
    const float* __restrict__ d, const float* __restrict__ S,
    const float* __restrict__ W, const float* __restrict__ sq,
    float* __restrict__ batchAcc) {
  __shared__ __align__(16) unsigned short lA[BM * LDSW];
  __shared__ __align__(16) unsigned short lB[BN * LDSW];

  const int b = blockIdx.z;
  const int rowBase = blockIdx.y * BM;
  const int colBase = blockIdx.x * BN;
  const int tid = threadIdx.x;
  const int lane = tid & 31;
  const int waveId = tid >> 5;
  const int wm = waveId & 3;
  const int wn = waveId >> 2;
  const int ln = lane & 15;
  const int h = lane >> 4;

  const float* dB = d + (size_t)b * NS * DD;

  v8f acc[2][4];
#pragma unroll
  for (int i = 0; i < 2; ++i)
#pragma unroll
    for (int j = 0; j < 4; ++j)
#pragma unroll
      for (int e = 0; e < 8; ++e) acc[i][j][e] = 0.f;

  for (int k0 = 0; k0 < DD; k0 += BK) {
#pragma unroll
    for (int t = 0; t < 4; ++t) {
      int idx = tid + t * 256;
      int r = idx >> 3;
      int c4 = (idx & 7) * 4;
      float4 av = *(const float4*)(dB + (size_t)(rowBase + r) * DD + k0 + c4);
      float4 bv = *(const float4*)(dB + (size_t)(colBase + r) * DD + k0 + c4);
      US4 au = {f2bf(av.x), f2bf(av.y), f2bf(av.z), f2bf(av.w)};
      US4 bu = {f2bf(bv.x), f2bf(bv.y), f2bf(bv.z), f2bf(bv.w)};
      *(US4*)(lA + r * LDSW + c4) = au;
      *(US4*)(lB + r * LDSW + c4) = bu;
    }
    __syncthreads();

    FragU bfr[4];
#pragma unroll
    for (int tn = 0; tn < 4; ++tn) {
      int n = wn * 64 + tn * 16 + ln;
      const unsigned short* pb = lB + n * LDSW + h * 16;
      bfr[tn].q[0] = *(const uint4*)(pb);
      bfr[tn].q[1] = *(const uint4*)(pb + 8);
    }
#pragma unroll
    for (int tm = 0; tm < 2; ++tm) {
      int m = wm * 32 + tm * 16 + ln;
      const unsigned short* pa = lA + m * LDSW;
      FragU af;
      af.q[0] = *(const uint4*)(pa + h * 16);
      af.q[1] = *(const uint4*)(pa + h * 16 + 8);
#pragma unroll
      for (int tn = 0; tn < 4; ++tn) {
        acc[tm][tn] = __builtin_amdgcn_wmma_f32_16x16x32_bf16(
            false, af.v, false, bfr[tn].v, (short)0, acc[tm][tn], false, false);
      }
    }
    __syncthreads();
  }

  epilogue(acc, b, rowBase, colBase, wm, wn, ln, h, lane, S, W, sq, batchAcc);
}

// ---------------- finalize: out = sum_b 2*sqrt(acc_b)/sqrt(denom) ----------------
__global__ void finalize(const float* __restrict__ ws, float* __restrict__ out) {
  float denom = sqrtf(ws[0]);
  float s = 0.f;
  for (int b = 0; b < NB; ++b) s += 2.f * sqrtf(ws[1 + b]) / denom;
  out[0] = s;
}

extern "C" void kernel_launch(void* const* d_in, const int* in_sizes, int n_in,
                              void* d_out, int out_size, void* d_ws, size_t ws_size,
                              hipStream_t stream) {
  const float* d = (const float*)d_in[0];  // [B, Ns, D]
  const float* S = (const float*)d_in[1];  // [Ns, Ns]
  const float* W = (const float*)d_in[2];  // [Ns, Ns]
  float* ws = (float*)d_ws;
  float* batchAcc = ws + 1;                     // ws[0]=denom, ws[1..64]=per-batch
  float* sq = ws + SQ_OFF;                      // [B*Ns] row squared norms
  unsigned short* dbf = (unsigned short*)((char*)d_ws + DBF_OFF);

  init_ws<<<1, 128, 0, stream>>>(ws);
  row_sq<<<(NB * NS) / 8, 256, 0, stream>>>(d, sq);
  denom_k<<<1024, 256, 0, stream>>>(W, ws);

  dim3 grid(NS / BN, NS / BM, NB);  // 8 x 8 x 64
  if (ws_size >= WS_NEED) {
    conv_bf16<<<(NB * NS * DD) / (8 * 256), 256, 0, stream>>>(d, dbf);
    gram_loss_bf16<<<grid, 256, 0, stream>>>(dbf, S, W, sq, batchAcc);
  } else {
    gram_loss_f32<<<grid, 256, 0, stream>>>(d, S, W, sq, batchAcc);
  }
  finalize<<<1, 1, 0, stream>>>(ws, (float*)d_out);
}
